// VectorQuantizer_8581344657812
// MI455X (gfx1250) — matrix operand order repaired and checked by reading, not compiled, not run on hardware
//
#include <hip/hip_runtime.h>

#define N_X   65536
#define DIM   256
#define K_CB  4096

typedef __attribute__((ext_vector_type(16))) _Float16 v16h;
typedef __attribute__((ext_vector_type(8)))  _Float16 v8h;
typedef __attribute__((ext_vector_type(8)))  float    v8f;

union V16HU { v16h v; uint4 u[2]; };

// ---------------------------------------------------------------------------
// Kernel 1: row L2-normalize. One wave (32 lanes) per row, 8 rows per block.
// Writes f32 normalized rows to dstF and f16 normalized rows to dstH.
// ---------------------------------------------------------------------------
__global__ __launch_bounds__(256) void vq_normalize_kernel(
    const float* __restrict__ src, float* __restrict__ dstF,
    _Float16* __restrict__ dstH, int nRows)
{
    const int wave = threadIdx.x >> 5;
    const int lane = threadIdx.x & 31;
    const int row  = blockIdx.x * 8 + wave;
    if (row >= nRows) return;

    const float4* p = (const float4*)(src + (size_t)row * DIM);
    float4 a = p[lane * 2 + 0];
    float4 b = p[lane * 2 + 1];

    float s = a.x*a.x + a.y*a.y + a.z*a.z + a.w*a.w
            + b.x*b.x + b.y*b.y + b.z*b.z + b.w*b.w;
    #pragma unroll
    for (int off = 1; off < 32; off <<= 1) s += __shfl_xor(s, off, 32);

    const float inv = 1.0f / fmaxf(sqrtf(s), 1e-12f);

    float4 oa = make_float4(a.x*inv, a.y*inv, a.z*inv, a.w*inv);
    float4 ob = make_float4(b.x*inv, b.y*inv, b.z*inv, b.w*inv);
    float4* q = (float4*)(dstF + (size_t)row * DIM);
    q[lane * 2 + 0] = oa;
    q[lane * 2 + 1] = ob;

    v8h h;
    h[0]=(_Float16)oa.x; h[1]=(_Float16)oa.y; h[2]=(_Float16)oa.z; h[3]=(_Float16)oa.w;
    h[4]=(_Float16)ob.x; h[5]=(_Float16)ob.y; h[6]=(_Float16)ob.z; h[7]=(_Float16)ob.w;
    *(v8h*)(dstH + (size_t)row * DIM + lane * 8) = h;
}

// ---------------------------------------------------------------------------
// Kernel 2: WMMA argmax of xn . cbn^T.
// 8 waves/block; each wave owns 16 x-rows (A resident in VGPRs all kernel).
// Per iteration the block stages 32 codebook rows in double-buffered LDS
// (one __syncthreads per iteration); each wave computes two 16x16 tiles with
// 4 independent v_wmma_f32_16x16x32_f16 chains of depth 4 (ILP for XDL pipe).
// ---------------------------------------------------------------------------
__global__ __launch_bounds__(256) void vq_argmax_wmma_kernel(
    const _Float16* __restrict__ xh, const _Float16* __restrict__ ch,
    int* __restrict__ idxOut)
{
    __shared__ __align__(16) _Float16 tileB[2][32 * DIM];   // 2 x 16 KB

    const int tid  = threadIdx.x;
    const int wave = tid >> 5;
    const int lane = tid & 31;
    const int m16  = lane & 15;
    const int rowBase = blockIdx.x * 128 + wave * 16;

    // ---- load A (16x256 f16) once: contiguous K (matches B) ----
    // lanes 0-15: elems 0-15 -> K c*32+0..15
    // lanes16-31: elems 0-15 -> K c*32+16..31
    const int kbA = (lane < 16) ? 0 : 16;
    const _Float16* xrow = xh + (size_t)(rowBase + m16) * DIM;
    V16HU A[8];
    #pragma unroll
    for (int c = 0; c < 8; ++c) {
        const int k0 = c * 32 + kbA;
        A[c].u[0] = *(const uint4*)(xrow + k0);
        A[c].u[1] = *(const uint4*)(xrow + k0 + 8);
    }

    float best[8];
    int   bidx[8];
    #pragma unroll
    for (int r = 0; r < 8; ++r) { best[r] = -3.0e38f; bidx[r] = 0; }

    // B layout (32x16 f16): lanes 0-15 hold K=c*32+0..15 contiguously,
    // lanes 16-31 hold K=c*32+16..31; column n = lane&15.
    const int kbB = (lane < 16) ? 0 : 16;

    auto loadTile = [&](int it, int buf) {
        // 32 rows x 256 f16 = 16 KB; 256 threads x 4 uint4
        const uint4* gsrc = (const uint4*)(ch + (size_t)it * 32 * DIM);
        uint4* l = (uint4*)(tileB[buf]);
        #pragma unroll
        for (int j = 0; j < 4; ++j) l[tid + 256 * j] = gsrc[tid + 256 * j];
    };

    constexpr int NIT = K_CB / 32;   // 128 iterations
    loadTile(0, 0);
    __syncthreads();

    for (int it = 0; it < NIT; ++it) {
        // prefetch next 32 rows into the other buffer (read-safe: that buffer
        // was last read in iteration it-1, fenced by the trailing barrier)
        if (it + 1 < NIT) loadTile(it + 1, (it + 1) & 1);

        const _Float16* base = tileB[it & 1];
        const _Float16* brow0 = base + m16 * DIM;          // cb rows it*32+ 0..15
        const _Float16* brow1 = base + (16 + m16) * DIM;   // cb rows it*32+16..31

        v8f a0 = {}, b0 = {}, a1 = {}, b1 = {};
        #pragma unroll
        for (int c = 0; c < 4; ++c) {
            V16HU B0, B0h, B1, B1h;
            B0.u[0]  = *(const uint4*)(brow0 + c * 32 + kbB);
            B0.u[1]  = *(const uint4*)(brow0 + c * 32 + kbB + 8);
            B0h.u[0] = *(const uint4*)(brow0 + (c + 4) * 32 + kbB);
            B0h.u[1] = *(const uint4*)(brow0 + (c + 4) * 32 + kbB + 8);
            B1.u[0]  = *(const uint4*)(brow1 + c * 32 + kbB);
            B1.u[1]  = *(const uint4*)(brow1 + c * 32 + kbB + 8);
            B1h.u[0] = *(const uint4*)(brow1 + (c + 4) * 32 + kbB);
            B1h.u[1] = *(const uint4*)(brow1 + (c + 4) * 32 + kbB + 8);
            a0 = __builtin_amdgcn_wmma_f32_16x16x32_f16(
                false, A[c].v,     false, B0.v,  (short)0, a0, false, false);
            b0 = __builtin_amdgcn_wmma_f32_16x16x32_f16(
                false, A[c + 4].v, false, B0h.v, (short)0, b0, false, false);
            a1 = __builtin_amdgcn_wmma_f32_16x16x32_f16(
                false, A[c].v,     false, B1.v,  (short)0, a1, false, false);
            b1 = __builtin_amdgcn_wmma_f32_16x16x32_f16(
                false, A[c + 4].v, false, B1h.v, (short)0, b1, false, false);
        }
        const v8f s0 = a0 + b0;
        const v8f s1 = a1 + b1;

        const int n0 = it * 32 + m16;
        const int n1 = n0 + 16;
        #pragma unroll
        for (int r = 0; r < 8; ++r) {
            if (s0[r] > best[r]) { best[r] = s0[r]; bidx[r] = n0; }
            if (s1[r] > best[r]) { best[r] = s1[r]; bidx[r] = n1; }
        }
        __syncthreads();   // everyone done reading buf[it&1] / writing buf[(it+1)&1]
    }

    // reduce argmax across the 16 lanes of each half (xor stays in half)
    #pragma unroll
    for (int r = 0; r < 8; ++r) {
        #pragma unroll
        for (int off = 1; off < 16; off <<= 1) {
            float ov = __shfl_xor(best[r], off, 32);
            int   oi = __shfl_xor(bidx[r], off, 32);
            if (ov > best[r] || (ov == best[r] && oi < bidx[r])) {
                best[r] = ov; bidx[r] = oi;
            }
        }
    }

    // C/D layout: VGPR r -> M=r (lanes 0-15) or M=8+r (lanes 16-31)
    if (m16 == 0) {
        const int mBase = rowBase + ((lane >> 4) * 8);
        #pragma unroll
        for (int r = 0; r < 8; ++r) idxOut[mBase + r] = bidx[r];
    }
}

// ---------------------------------------------------------------------------
// Kernel 3: gather quantized rows, overwrite xn in d_out, accumulate loss.
// 64 threads per row (float4 each); one atomic add per block.
// ---------------------------------------------------------------------------
__global__ __launch_bounds__(256) void vq_output_kernel(
    const int* __restrict__ idx, const float* __restrict__ cbF,
    float* __restrict__ dq, float* __restrict__ outIdx,
    float* __restrict__ acc)
{
    __shared__ float red[8];
    const int gid = blockIdx.x * 256 + threadIdx.x;
    const int row = gid >> 6;
    const int col = (gid & 63) * 4;

    const int id = idx[row];
    float4 q = *(const float4*)(cbF + (size_t)id * DIM + col);
    float4 x = *(const float4*)(dq  + (size_t)row * DIM + col);
    const float dx = q.x - x.x, dy = q.y - x.y, dz = q.z - x.z, dw = q.w - x.w;
    float s = dx*dx + dy*dy + dz*dz + dw*dw;
    *(float4*)(dq + (size_t)row * DIM + col) = q;

    if ((gid & 63) == 0) outIdx[row] = (float)id;

    #pragma unroll
    for (int off = 1; off < 32; off <<= 1) s += __shfl_xor(s, off, 32);
    const int lane = threadIdx.x & 31, wave = threadIdx.x >> 5;
    if (lane == 0) red[wave] = s;
    __syncthreads();
    if (threadIdx.x < 8) {
        s = red[threadIdx.x];
        #pragma unroll
        for (int off = 1; off < 8; off <<= 1) s += __shfl_xor(s, off, 32);
        if (threadIdx.x == 0) atomicAdd(acc, s);
    }
}

__global__ void vq_zero_acc(float* acc) { *acc = 0.0f; }

__global__ void vq_finalize(const float* acc, float* lossOut)
{
    *lossOut = 1.25f * (*acc) / (float)((size_t)N_X * DIM);
}

// ---------------------------------------------------------------------------
extern "C" void kernel_launch(void* const* d_in, const int* in_sizes, int n_in,
                              void* d_out, int out_size, void* d_ws, size_t ws_size,
                              hipStream_t stream)
{
    (void)in_sizes; (void)n_in; (void)out_size; (void)ws_size;
    const float* x  = (const float*)d_in[0];
    const float* cb = (const float*)d_in[1];

    float* q       = (float*)d_out;                 // [N_X, DIM] quantized_st
    float* lossOut = q + (size_t)N_X * DIM;         // [1]
    float* idxOut  = lossOut + 1;                   // [N_X] idx as float

    // workspace layout
    char* w = (char*)d_ws;
    _Float16* xh  = (_Float16*)w;                                   // 32 MB
    _Float16* chh = (_Float16*)(w + (size_t)N_X * DIM * 2);         //  2 MB
    float*    cbF = (float*)(w + (size_t)N_X * DIM * 2
                               + (size_t)K_CB * DIM * 2);           //  4 MB
    int*      idxW = (int*)((char*)cbF + (size_t)K_CB * DIM * 4);   // 256 KB
    float*    acc  = (float*)(idxW + N_X);                          //   4 B

    // 1) normalize x -> (d_out f32 slot, f16 ws) and codebook -> (f32 ws, f16 ws)
    vq_normalize_kernel<<<N_X / 8, 256, 0, stream>>>(x, q, xh, N_X);
    vq_normalize_kernel<<<K_CB / 8, 256, 0, stream>>>(cb, cbF, chh, K_CB);

    // 2) WMMA argmax over codebook
    vq_argmax_wmma_kernel<<<N_X / 128, 256, 0, stream>>>(xh, chh, idxW);

    // 3) gather + loss
    vq_zero_acc<<<1, 1, 0, stream>>>(acc);
    vq_output_kernel<<<(size_t)N_X * DIM / 1024, 256, 0, stream>>>(idxW, cbF, q, idxOut, acc);
    vq_finalize<<<1, 1, 0, stream>>>(acc, lossOut);
}
